// SelectiveStateSpaceMixer_39101382262865
// MI455X (gfx1250) — matrix operand order repaired and checked by reading, not compiled, not run on hardware
//
#include <hip/hip_runtime.h>
#include <hip/hip_bf16.h>

// ---------------------------------------------------------------------------
// CDNA5 async global->LDS support (probe-verified on this toolchain:
// first arg `int __vector(4) addrspace(1)*`).
// ---------------------------------------------------------------------------
#if __has_builtin(__builtin_amdgcn_global_load_async_to_lds_b128) && \
    __has_builtin(__builtin_amdgcn_s_wait_asynccnt)
#define HAVE_ASYNC_LDS 1
#else
#define HAVE_ASYNC_LDS 0
#endif

#if __has_builtin(__builtin_amdgcn_sched_barrier)
#define SCHED_FENCE() __builtin_amdgcn_sched_barrier(0)
#else
#define SCHED_FENCE()
#endif

typedef __attribute__((ext_vector_type(16))) __bf16 v16bf;
typedef __attribute__((ext_vector_type(8)))  float  v8f;
typedef int v4i __attribute__((vector_size(16)));
typedef __attribute__((address_space(1))) v4i* gv4i_ptr;
typedef __attribute__((address_space(3))) v4i* lv4i_ptr;

union FragAB { v16bf v; uint4 q[2]; };

__device__ __forceinline__ unsigned short f2bf(float f) {
  unsigned int u = __float_as_uint(f);
  u += 0x7fffu + ((u >> 16) & 1u);          // round to nearest even
  return (unsigned short)(u >> 16);
}
__device__ __forceinline__ float bf2f(unsigned short h) {
  return __uint_as_float(((unsigned int)h) << 16);
}
__device__ __forceinline__ float sigm(float x) { return 1.0f / (1.0f + __expf(-x)); }

// ---------------------------------------------------------------------------
// bf16 WMMA GEMM with LDS double buffering + async global->LDS pipeline.
// A  : M x K bf16 row-major (lda = K)
// BT : N x K bf16 row-major (pre-transposed weights, ldb = K)
//  EPI 0: split: cols < N/2 -> bf16 x_main ; cols >= N/2 -> bf16 sigmoid(gate)
//  EPI 1: dt/B/C: sigmoid on first 64 cols, fp32 out (row stride N)
//  EPI 2: v = (acc + bias + D*u) * sgate -> bf16
//  EPI 3: plain fp32 out + bias
// Per-wave pipeline (order pinned by sched_barrier):
//   wait(buf p writes) -> barrier -> frag ds-loads(p) -> barrier ->
//   async issue(k+1 -> p^1) -> WMMA(k)
// ---------------------------------------------------------------------------
template<int BM, int BN, int EPI>
__global__ __launch_bounds__(256)
void gemm_bf16_ker(const unsigned short* __restrict__ A, int lda,
                   const unsigned short* __restrict__ BT, int ldb,
                   const float* __restrict__ bias,
                   int M, int N, int Kd,
                   unsigned short* __restrict__ out_bf_a,
                   unsigned short* __restrict__ out_bf_b,
                   float* __restrict__ out_f,
                   const float* __restrict__ Dvec,
                   const float* __restrict__ u_f32,
                   const unsigned short* __restrict__ sgate)
{
  constexpr int WN   = BN / 64;          // waves along N
  constexpr int WM   = 8 / WN;           // waves along M (8 waves / block)
  constexpr int MSUB = BM / (WM * 16);   // 16x16 subtiles per wave along M
  constexpr int LD   = 40;               // 32 shorts + 8 pad (80B rows)
  constexpr int TILE = (BM + BN) * LD;   // shorts per buffer
  constexpr int AIT  = (BM * 4) / 256;   // b128 copies per thread, A tile
  constexpr int BIT  = (BN * 4) / 256;   // b128 copies per thread, B tile

  __shared__ unsigned short smem[2 * TILE];

  const int tid  = threadIdx.x;
  const int lane = tid & 31;
  const int wv   = tid >> 5;
  const int wm   = wv % WM;
  const int wn   = wv / WM;
  const int mOff = wm * (MSUB * 16);
  const int nOff = wn * 64;
  const int r    = lane & 15;
  const int hi   = lane >> 4;            // 0 or 1 (lane half)

  const int blockM = blockIdx.y * BM;
  const int blockN = blockIdx.x * BN;

  // ---- per-thread copy descriptors (constant across k-chunks) ----
  const int crow = tid >> 2;             // 0..63
  const int cc   = (tid & 3) * 8;        // short offset within 32-short row
  const unsigned short* aSrc[AIT]; int aDst[AIT];
#pragma unroll
  for (int it = 0; it < AIT; it++) {
    int row = crow + it * 64;
    aSrc[it] = A + (size_t)(blockM + row) * lda + cc;
    aDst[it] = row * LD + cc;
  }
  const unsigned short* bSrc[BIT]; int bDst[BIT];
#pragma unroll
  for (int it = 0; it < BIT; it++) {
    int row = crow + it * 64;
    bSrc[it] = BT + (size_t)(blockN + row) * ldb + cc;
    bDst[it] = BM * LD + row * LD + cc;
  }

  auto issue_tile = [&](int k0, int buf) {
    unsigned short* s = smem + buf * TILE;
#pragma unroll
    for (int it = 0; it < AIT; it++) {
#if HAVE_ASYNC_LDS
      __builtin_amdgcn_global_load_async_to_lds_b128(
          (gv4i_ptr)(aSrc[it] + k0), (lv4i_ptr)(s + aDst[it]), 0, 0);
#else
      *(uint4*)(s + aDst[it]) = *(const uint4*)(aSrc[it] + k0);
#endif
    }
#pragma unroll
    for (int it = 0; it < BIT; it++) {
#if HAVE_ASYNC_LDS
      __builtin_amdgcn_global_load_async_to_lds_b128(
          (gv4i_ptr)(bSrc[it] + k0), (lv4i_ptr)(s + bDst[it]), 0, 0);
#else
      *(uint4*)(s + bDst[it]) = *(const uint4*)(bSrc[it] + k0);
#endif
    }
  };

  v8f zero = {0.f, 0.f, 0.f, 0.f, 0.f, 0.f, 0.f, 0.f};
  v8f acc[MSUB][4];
#pragma unroll
  for (int mi = 0; mi < MSUB; mi++)
#pragma unroll
    for (int ni = 0; ni < 4; ni++) acc[mi][ni] = zero;

  const int nk = Kd / 32;
  issue_tile(0, 0);                       // prologue: tile 0 -> buffer 0

  for (int kt = 0; kt < nk; kt++) {
    const int p = kt & 1;
    SCHED_FENCE();                        // keep this wait below prior WMMAs
#if HAVE_ASYNC_LDS
    __builtin_amdgcn_s_wait_asynccnt(0);  // writes into buffer p complete
#endif
    __syncthreads();                      // visible to all waves

    const unsigned short* sA = smem + p * TILE;
    const unsigned short* sB = sA + BM * LD;

    // ---- fragments (CDNA5 16-bit WMMA lane layouts) ----
    FragAB fa[MSUB], fb[4];
#pragma unroll
    for (int mi = 0; mi < MSUB; mi++) {
      // A 16x32: lanes 0-15 row M=lane, K 0..7 & 16..23 ; lanes 16-31 K 8..15 & 24..31
      const unsigned short* pp = &sA[(mOff + mi * 16 + r) * LD + hi * 8];
      fa[mi].q[0] = *(const uint4*)pp;
      fa[mi].q[1] = *(const uint4*)(pp + 16);
    }
#pragma unroll
    for (int ni = 0; ni < 4; ni++) {
      // B 32x16: lanes 0-15 col N=lane, K 0..7 & 16..23 ; lanes 16-31 K 8..15 & 24..31
      const unsigned short* pp = &sB[(nOff + ni * 16 + r) * LD + hi * 8];
      fb[ni].q[0] = *(const uint4*)pp;
      fb[ni].q[1] = *(const uint4*)(pp + 16);
    }
    __syncthreads();                      // all waves done reading buffer p

    if (kt + 1 < nk) issue_tile((kt + 1) * 32, p ^ 1);  // overlap with WMMA
    SCHED_FENCE();                        // keep WMMAs below the async issue

#pragma unroll
    for (int mi = 0; mi < MSUB; mi++)
#pragma unroll
      for (int ni = 0; ni < 4; ni++)
        acc[mi][ni] = __builtin_amdgcn_wmma_f32_16x16x32_bf16(
            false, fa[mi].v, false, fb[ni].v, (short)0, acc[mi][ni], false, false);
  }

  // ---- epilogue: C layout -> VGPR e: lanes 0-15 M=e, lanes 16-31 M=e+8 ----
#pragma unroll
  for (int mi = 0; mi < MSUB; mi++) {
#pragma unroll
    for (int ni = 0; ni < 4; ni++) {
#pragma unroll
      for (int e = 0; e < 8; e++) {
        int row = blockM + mOff + mi * 16 + e + hi * 8;
        int col = blockN + nOff + ni * 16 + r;
        float v = acc[mi][ni][e] + bias[col];
        if (EPI == 0) {
          int h = N >> 1;
          if (col < h) out_bf_a[(size_t)row * h + col] = f2bf(v);
          else         out_bf_b[(size_t)row * h + (col - h)] = f2bf(sigm(v));
        } else if (EPI == 1) {
          if (col < 64) v = sigm(v);
          out_f[(size_t)row * N + col] = v;
        } else if (EPI == 2) {
          v += Dvec[col] * u_f32[(size_t)row * N + col];
          v *= bf2f(sgate[(size_t)row * N + col]);
          out_bf_a[(size_t)row * N + col] = f2bf(v);
        } else {
          out_f[(size_t)row * N + col] = v;
        }
      }
    }
  }
}

// ---------------------------------------------------------------------------
// fp32 (KxN) -> bf16 transposed (NxK), 32x32 LDS tiles, both sides coalesced.
// ---------------------------------------------------------------------------
__global__ __launch_bounds__(256)
void transpose_bf16_ker(const float* __restrict__ src, unsigned short* __restrict__ dst,
                        int K, int N) {
  __shared__ unsigned short t[32][33];
  int kb = blockIdx.y * 32, nb = blockIdx.x * 32;
  int tx = threadIdx.x & 31, ty = threadIdx.x >> 5;   // 32 x 8
#pragma unroll
  for (int j = 0; j < 32; j += 8)
    t[ty + j][tx] = f2bf(src[(size_t)(kb + ty + j) * N + nb + tx]);
  __syncthreads();
#pragma unroll
  for (int j = 0; j < 32; j += 8)
    dst[(size_t)(nb + ty + j) * K + kb + tx] = t[tx][ty + j];
}

// fp32 -> bf16 (no transpose), for activations x
__global__ void cvt_bf16_ker(const float* __restrict__ s, unsigned short* __restrict__ d, int n) {
  int i = blockIdx.x * 256 + threadIdx.x;
  if (i < n) d[i] = f2bf(s[i]);
}

// Pack W_dt|W_B|W_C (each 2048x64) -> transposed 192x2048 bf16
__global__ void pack_wdtbc_ker(const float* __restrict__ Wdt, const float* __restrict__ Wb,
                               const float* __restrict__ Wc, unsigned short* __restrict__ dst) {
  int i = blockIdx.x * 256 + threadIdx.x;   // over 2048*192, read-coalesced
  if (i >= 2048 * 192) return;
  int k = i / 192, c = i % 192;
  float v = (c < 64) ? Wdt[k * 64 + c]
          : (c < 128) ? Wb[k * 64 + (c - 64)]
                      : Wc[k * 64 + (c - 128)];
  dst[(size_t)c * 2048 + k] = f2bf(v);
}
__global__ void pack_bdtbc_ker(const float* __restrict__ bdt, const float* __restrict__ bb,
                               const float* __restrict__ bc, float* __restrict__ dst) {
  int c = threadIdx.x;  // 192 threads
  dst[c] = (c < 64) ? bdt[c] : (c < 128) ? bb[c - 64] : bc[c - 128];
}

// ---------------------------------------------------------------------------
// Depthwise causal conv (K=3) + bias + SiLU
// ---------------------------------------------------------------------------
__global__ __launch_bounds__(256)
void conv_ker(const unsigned short* __restrict__ xm,
              const float* __restrict__ cw, const float* __restrict__ cb,
              float* __restrict__ uf, unsigned short* __restrict__ ub, int S) {
  size_t idx = (size_t)blockIdx.x * 256 + threadIdx.x;  // over MB*2048
  int c = (int)(idx & 2047);
  size_t row = idx >> 11;
  int t = (int)(row % S);
  float x2 = bf2f(xm[idx]);
  float x1 = (t >= 1) ? bf2f(xm[idx - 2048]) : 0.f;
  float x0 = (t >= 2) ? bf2f(xm[idx - 4096]) : 0.f;
  float a = cw[c * 3 + 0] * x0 + cw[c * 3 + 1] * x1 + cw[c * 3 + 2] * x2 + cb[c];
  float r = a * sigm(a);
  uf[idx] = r;
  ub[idx] = f2bf(r);
}

// ---------------------------------------------------------------------------
// Parallel scan: s_t = (1-dt)*s_{t-1} + dt*B_t ; y_t = C_t * s_t
// One block per (batch, state) chain; chunked Hillis-Steele over (A,B) ops.
// ---------------------------------------------------------------------------
__global__ __launch_bounds__(256)
void scan_ker(const float* __restrict__ dtBC, float* __restrict__ y, int S) {
  __shared__ float sA[256], sB[256];
  const int chain = blockIdx.x;   // 0..127
  const int b = chain >> 6, n = chain & 63;
  const int t = threadIdx.x;
  const int chunk = S / 256;      // 16
  const size_t base = (size_t)b * S;

  float Ac = 1.f, Bc = 0.f;
  for (int i = 0; i < chunk; i++) {
    size_t rowi = base + (size_t)t * chunk + i;
    float d  = dtBC[rowi * 192 + n];
    float bb = d * dtBC[rowi * 192 + 64 + n];
    float a  = 1.f - d;
    Ac = a * Ac;
    Bc = a * Bc + bb;
  }
  sA[t] = Ac; sB[t] = Bc;
  __syncthreads();
  for (int off = 1; off < 256; off <<= 1) {
    float cA = sA[t], cB = sB[t], pA = 1.f, pB = 0.f;
    if (t >= off) { pA = sA[t - off]; pB = sB[t - off]; }
    __syncthreads();
    sA[t] = cA * pA;
    sB[t] = cA * pB + cB;
    __syncthreads();
  }
  float s = (t == 0) ? 0.f : sB[t - 1];
  for (int i = 0; i < chunk; i++) {
    size_t rowi = base + (size_t)t * chunk + i;
    float d  = dtBC[rowi * 192 + n];
    float bb = d * dtBC[rowi * 192 + 64 + n];
    float c  = dtBC[rowi * 192 + 128 + n];
    s = (1.f - d) * s + bb;
    y[rowi * 64 + n] = c * s;
  }
}

// ---------------------------------------------------------------------------
// LayerNorm over 64 states, one wave32 per row (2 elems/lane), bf16 out.
// ---------------------------------------------------------------------------
__global__ __launch_bounds__(256)
void ln_ker(const float* __restrict__ y, unsigned short* __restrict__ yln) {
  int wv = threadIdx.x >> 5, lane = threadIdx.x & 31;
  int row = blockIdx.x * 8 + wv;
  const float* p = y + (size_t)row * 64;
  float v0 = p[lane], v1 = p[lane + 32];
  float s = v0 + v1, sq = v0 * v0 + v1 * v1;
#pragma unroll
  for (int m = 16; m >= 1; m >>= 1) {
    s  += __shfl_xor(s,  m, 32);
    sq += __shfl_xor(sq, m, 32);
  }
  float mu  = s * (1.f / 64.f);
  float var = sq * (1.f / 64.f) - mu * mu;
  float inv = rsqrtf(var + 1e-5f);
  unsigned short* o = yln + (size_t)row * 64;
  o[lane]      = f2bf((v0 - mu) * inv);
  o[lane + 32] = f2bf((v1 - mu) * inv);
}

// ---------------------------------------------------------------------------
// Launch
// ---------------------------------------------------------------------------
extern "C" void kernel_launch(void* const* d_in, const int* in_sizes, int n_in,
                              void* d_out, int out_size, void* d_ws, size_t ws_size,
                              hipStream_t stream) {
  const float* x    = (const float*)d_in[0];
  const float* Win  = (const float*)d_in[1];
  const float* bin  = (const float*)d_in[2];
  const float* cw   = (const float*)d_in[3];
  const float* cb   = (const float*)d_in[4];
  const float* Wdt  = (const float*)d_in[5];
  const float* bdt  = (const float*)d_in[6];
  const float* Wb   = (const float*)d_in[7];
  const float* bb   = (const float*)d_in[8];
  const float* Wc   = (const float*)d_in[9];
  const float* bc   = (const float*)d_in[10];
  const float* Wsi  = (const float*)d_in[11];
  const float* bsi  = (const float*)d_in[12];
  const float* Dv   = (const float*)d_in[13];
  const float* Wout = (const float*)d_in[14];
  const float* bout = (const float*)d_in[15];
  float* out = (float*)d_out;

  constexpr int S  = 4096;
  constexpr int MB = 2 * S;          // 8192 rows
  char* w = (char*)d_ws;
  size_t off = 0;
  auto alloc = [&](size_t bytes) {
    void* p = w + off;
    off += (bytes + 255) & ~(size_t)255;
    return p;
  };
  unsigned short* xbf     = (unsigned short*)alloc((size_t)MB * 1024 * 2);
  unsigned short* winT    = (unsigned short*)alloc((size_t)1024 * 4096 * 2);   // 4096x1024
  unsigned short* wdtbcT  = (unsigned short*)alloc((size_t)2048 * 192 * 2);    // 192x2048
  float*          bdtbc   = (float*)         alloc((size_t)192 * 4);
  unsigned short* wsiT    = (unsigned short*)alloc((size_t)64 * 2048 * 2);     // 2048x64
  unsigned short* woutT   = (unsigned short*)alloc((size_t)2048 * 1024 * 2);   // 1024x2048
  unsigned short* xmainbf = (unsigned short*)alloc((size_t)MB * 2048 * 2);
  unsigned short* sgatebf = (unsigned short*)alloc((size_t)MB * 2048 * 2);
  float*          uf      = (float*)         alloc((size_t)MB * 2048 * 4);
  unsigned short* ubf     = (unsigned short*)alloc((size_t)MB * 2048 * 2);
  float*          dtbc    = (float*)         alloc((size_t)MB * 192 * 4);
  float*          ys      = (float*)         alloc((size_t)MB * 64 * 4);
  unsigned short* ylnbf   = (unsigned short*)alloc((size_t)MB * 64 * 2);
  unsigned short* vbf     = (unsigned short*)alloc((size_t)MB * 2048 * 2);
  (void)ws_size; (void)in_sizes; (void)n_in; (void)out_size;

  // 1. precision conversion: weights -> bf16 transposed (N-major), x -> bf16
  cvt_bf16_ker<<<(MB * 1024 + 255) / 256, 256, 0, stream>>>(x, xbf, MB * 1024);
  transpose_bf16_ker<<<dim3(4096 / 32, 1024 / 32), 256, 0, stream>>>(Win, winT, 1024, 4096);
  transpose_bf16_ker<<<dim3(2048 / 32, 64 / 32), 256, 0, stream>>>(Wsi, wsiT, 64, 2048);
  transpose_bf16_ker<<<dim3(1024 / 32, 2048 / 32), 256, 0, stream>>>(Wout, woutT, 2048, 1024);
  pack_wdtbc_ker<<<(2048 * 192 + 255) / 256, 256, 0, stream>>>(Wdt, Wb, Wc, wdtbcT);
  pack_bdtbc_ker<<<1, 192, 0, stream>>>(bdt, bb, bc, bdtbc);

  // 2. xz = x @ W_in + b_in -> x_main bf16 | sigmoid(gate) bf16
  gemm_bf16_ker<128, 128, 0><<<dim3(4096 / 128, MB / 128), 256, 0, stream>>>(
      xbf, 1024, winT, 1024, bin, MB, 4096, 1024,
      xmainbf, sgatebf, nullptr, nullptr, nullptr, nullptr);

  // 3. depthwise causal conv + SiLU -> u
  conv_ker<<<(MB * 2048) / 256, 256, 0, stream>>>(xmainbf, cw, cb, uf, ubf, S);

  // 4. [dt|B|C] = u @ [W_dt|W_B|W_C] + bias (sigmoid on dt block)
  gemm_bf16_ker<128, 64, 1><<<dim3(192 / 64, MB / 128), 256, 0, stream>>>(
      ubf, 2048, wdtbcT, 2048, bdtbc, MB, 192, 2048,
      nullptr, nullptr, dtbc, nullptr, nullptr, nullptr);

  // 5. selective scan + layernorm
  scan_ker<<<128, 256, 0, stream>>>(dtbc, ys, S);
  ln_ker<<<MB / 8, 256, 0, stream>>>(ys, ylnbf);

  // 6. v = (yln @ W_si + b_si + D*u) * sigmoid(gate)
  gemm_bf16_ker<128, 128, 2><<<dim3(2048 / 128, MB / 128), 256, 0, stream>>>(
      ylnbf, 64, wsiT, 64, bsi, MB, 2048, 64,
      vbf, nullptr, nullptr, Dv, uf, sgatebf);

  // 7. out = v @ W_out + b_out
  gemm_bf16_ker<128, 128, 3><<<dim3(1024 / 128, MB / 128), 256, 0, stream>>>(
      vbf, 2048, woutT, 2048, bout, MB, 1024, 2048,
      nullptr, nullptr, out, nullptr, nullptr, nullptr);
}
